// semidual_26010321944637
// MI455X (gfx1250) — matrix operand order repaired and checked by reading, not compiled, not run on hardware
//
#include <hip/hip_runtime.h>

// ---------------------------------------------------------------------------
// semidual OT loss:  mean_i min_j (|x_i|^2 + |y_j|^2 - 2 x_i.y_j - psi_j) + mean(psi)
// Fused bf16x3 WMMA GEMM + row-min.  N = M = 16384, D = 512 (f32 inputs).
// Data path: one-time f32 -> (bf16 hi, bf16 lo) plane split in workspace,
// then Tensor Data Mover (tensor_load_to_lds) streams tiles into LDS.
// ---------------------------------------------------------------------------

#define DDIM   512
#define BM     128
#define BN     128
#define BK     32
#define NCHUNK (DDIM / BK)      // 16
#define ROWU4  5                // padded LDS row: 40 bf16 = 80 B = 5 uint4

#if (__has_builtin(__builtin_amdgcn_tensor_load_to_lds) && \
     __has_builtin(__builtin_amdgcn_s_wait_tensorcnt))
#define HAVE_TDM 1
#else
#define HAVE_TDM 0
#endif

typedef __attribute__((ext_vector_type(16))) __bf16    v16bf;
typedef __attribute__((ext_vector_type(8)))  float     v8f;
typedef __attribute__((ext_vector_type(4)))  unsigned  u32x4;
typedef __attribute__((ext_vector_type(8)))  int       i32x8;
typedef __attribute__((ext_vector_type(4)))  int       i32x4;

// ---- helpers --------------------------------------------------------------

static __device__ __forceinline__ unsigned short f2bf(float f) {
    unsigned u = __float_as_uint(f);
    unsigned r = u + 0x7FFFu + ((u >> 16) & 1u);   // round-to-nearest-even
    return (unsigned short)(r >> 16);
}
static __device__ __forceinline__ float bf2f(unsigned short h) {
    return __uint_as_float(((unsigned)h) << 16);
}
static __device__ __forceinline__ unsigned pack2(unsigned short a, unsigned short b) {
    return (unsigned)a | ((unsigned)b << 16);
}
// order-preserving float <-> uint mapping (monotonic => integer min == float min)
static __device__ __forceinline__ unsigned f2ord(float f) {
    unsigned u = __float_as_uint(f);
    return (u & 0x80000000u) ? ~u : (u | 0x80000000u);
}
static __device__ __forceinline__ float ord2f(unsigned u) {
    unsigned b = (u & 0x80000000u) ? (u ^ 0x80000000u) : ~u;
    return __uint_as_float(b);
}
static __device__ __forceinline__ v16bf to_v16bf(uint4 a, uint4 b) {
    union { uint4 q[2]; v16bf v; } u;
    u.q[0] = a; u.q[1] = b;
    return u.v;
}

// shared 24-WMMA K-chunk (bf16x3: hi*hi + hi*lo + lo*hi)
static __device__ __forceinline__ void wmma_chunk(
    const uint4* __restrict__ xh, const uint4* __restrict__ xl,
    const uint4* __restrict__ yh, const uint4* __restrict__ yl,
    int wm, int wn, int lrow, int lhalf, v8f (&acc)[4][2])
{
    v16bf ahi[4], alo[4], bhi[2], blo[2];
    // A fragment (16x32 bf16): lane<16 -> K0..15, lane>=16 -> K16..31
#pragma unroll
    for (int mt = 0; mt < 4; ++mt) {
        int ab = (wm * 64 + mt * 16 + lrow) * ROWU4 + lhalf * 2;
        ahi[mt] = to_v16bf(xh[ab], xh[ab + 1]);
        alo[mt] = to_v16bf(xl[ab], xl[ab + 1]);
    }
    // B fragment (32x16 bf16): lane<16 -> K0..15, lane>=16 -> K16..31 of column n
#pragma unroll
    for (int nt = 0; nt < 2; ++nt) {
        int bb = (wn * 32 + nt * 16 + lrow) * ROWU4 + lhalf * 2;
        bhi[nt] = to_v16bf(yh[bb], yh[bb + 1]);
        blo[nt] = to_v16bf(yl[bb], yl[bb + 1]);
    }
#pragma unroll
    for (int mt = 0; mt < 4; ++mt)
#pragma unroll
        for (int nt = 0; nt < 2; ++nt) {
            acc[mt][nt] = __builtin_amdgcn_wmma_f32_16x16x32_bf16(
                false, ahi[mt], false, bhi[nt], (short)0, acc[mt][nt], false, false);
            acc[mt][nt] = __builtin_amdgcn_wmma_f32_16x16x32_bf16(
                false, ahi[mt], false, blo[nt], (short)0, acc[mt][nt], false, false);
            acc[mt][nt] = __builtin_amdgcn_wmma_f32_16x16x32_bf16(
                false, alo[mt], false, bhi[nt], (short)0, acc[mt][nt], false, false);
        }
}

// epilogue: val = t[j] - 2*dot ; per-row min across columns -> smin -> rowmin
static __device__ __forceinline__ void min_epilogue(
    v8f (&acc)[4][2], const float* __restrict__ T, unsigned* __restrict__ rowmin,
    unsigned* smin, int rowM, int rowN, int tid, int wm, int wn, int lrow, int lhalf)
{
    float t0 = T[rowN + wn * 32 + 0  + lrow];
    float t1 = T[rowN + wn * 32 + 16 + lrow];
#pragma unroll
    for (int mt = 0; mt < 4; ++mt) {
#pragma unroll
        for (int r = 0; r < 8; ++r) {
            float v = fminf(fmaf(-2.f, acc[mt][0][r], t0),
                            fmaf(-2.f, acc[mt][1][r], t1));
#pragma unroll
            for (int m = 1; m <= 8; m <<= 1) v = fminf(v, __shfl_xor(v, m, 32));
            if (lrow == 0) {
                int rloc = wm * 64 + mt * 16 + lhalf * 8 + r;   // C layout: lanes16-31 -> M+8
                atomicMin(&smin[rloc], f2ord(v));               // ds_min_u32
            }
        }
    }
    __syncthreads();
    if (tid < BM) atomicMin(&rowmin[rowM + tid], smin[tid]);    // global_atomic_min_u32
}

// ---- kernel 1: x2[i], t[j] = y2[j]-psi[j], rowmin init --------------------
__global__ __launch_bounds__(256) void prep_kernel(
    const float* __restrict__ X, const float* __restrict__ Y,
    const float* __restrict__ PSI,
    float* __restrict__ x2, float* __restrict__ t,
    unsigned* __restrict__ rowmin, int N, int M)
{
    const int gw   = blockIdx.x * 8 + (threadIdx.x >> 5);
    const int lane = threadIdx.x & 31;
    if (gw >= N + M) return;

    const float* p = (gw < N) ? (X + (size_t)gw * DDIM)
                              : (Y + (size_t)(gw - N) * DDIM);
    float s = 0.f;
#pragma unroll
    for (int j = 0; j < DDIM / 32; ++j) {
        float v = p[lane + 32 * j];
        s = fmaf(v, v, s);
    }
#pragma unroll
    for (int m = 16; m >= 1; m >>= 1) s += __shfl_xor(s, m, 32);

    if (lane == 0) {
        if (gw < N) { x2[gw] = s; rowmin[gw] = 0xFFFFFFFFu; }
        else        { t[gw - N] = s - PSI[gw - N]; }
    }
}

// ---- kernel 2a: one-time f32 -> bf16 hi/lo plane split --------------------
__global__ __launch_bounds__(256) void convert_kernel(
    const float* __restrict__ X, const float* __restrict__ Y,
    unsigned short* __restrict__ Xh, unsigned short* __restrict__ Xl,
    unsigned short* __restrict__ Yh, unsigned short* __restrict__ Yl, int N)
{
    const size_t nx   = (size_t)N * DDIM;
    size_t base = ((size_t)blockIdx.x * 256 + threadIdx.x) * 8;
    const float* src; unsigned short *h, *l; size_t off;
    if (base < nx) { src = X; h = Xh; l = Xl; off = base; }
    else           { src = Y; h = Yh; l = Yl; off = base - nx; }

    float4 a = ((const float4*)(src + off))[0];
    float4 b = ((const float4*)(src + off))[1];
    float v[8] = {a.x, a.y, a.z, a.w, b.x, b.y, b.z, b.w};
    unsigned hq[4], lq[4];
#pragma unroll
    for (int e = 0; e < 4; ++e) {
        unsigned short h0 = f2bf(v[2*e]), h1 = f2bf(v[2*e+1]);
        unsigned short l0 = f2bf(v[2*e]   - bf2f(h0));
        unsigned short l1 = f2bf(v[2*e+1] - bf2f(h1));
        hq[e] = pack2(h0, h1); lq[e] = pack2(l0, l1);
    }
    *(uint4*)(h + off) = make_uint4(hq[0], hq[1], hq[2], hq[3]);
    *(uint4*)(l + off) = make_uint4(lq[0], lq[1], lq[2], lq[3]);
}

#if HAVE_TDM
// ---- TDM: issue one 2D bf16 tile (BMxBK elems) global -> padded LDS -------
static __device__ __forceinline__ void tdm_load_tile(
    const unsigned short* g, unsigned lds_byte_addr)
{
    unsigned long long ga = (unsigned long long)(uintptr_t)g;
    u32x4 g0;
    g0[0] = 1u;                                   // count=1, user descriptor
    g0[1] = lds_byte_addr;                        // lds_addr
    g0[2] = (unsigned)ga;                         // global_addr[31:0]
    g0[3] = (unsigned)(ga >> 32) | (2u << 30);    // global_addr[56:32] | type=2
    i32x8 g1;
    // pad_amount=3(4 dw), pad_interval=3(16 dw), pad_en, data_size=1(2B), wg_mask=0
    g1[0] = (int)((3u << 25) | (3u << 22) | (1u << 20) | (1u << 16));
    g1[1] = (int)((unsigned)(DDIM & 0xFFFF) << 16);        // tensor_dim0 = 512
    g1[2] = (int)((16384u & 0xFFFFu) << 16);               // tensor_dim1 = 16384
    g1[3] = (int)((unsigned)BK << 16);                     // tile_dim0 = 32
    g1[4] = (int)(unsigned)BM;                             // tile_dim1 = 128, tile_dim2 = 0
    g1[5] = (int)(unsigned)DDIM;                           // tensor_dim0_stride = 512
    g1[6] = 0;
    g1[7] = 0;
    i32x4 g2 = {0, 0, 0, 0};
    i32x4 g3 = {0, 0, 0, 0};
    i32x8 g4 = {0, 0, 0, 0, 0, 0, 0, 0};          // trailing group (6-arg form)
    __builtin_amdgcn_tensor_load_to_lds(g0, g1, g2, g3, g4, 0);
}

// ---- kernel 2b: fused WMMA GEMM + row-min, TDM-fed ------------------------
__global__ __launch_bounds__(256) void gemm_min_tdm_kernel(
    const unsigned short* __restrict__ Xh, const unsigned short* __restrict__ Xl,
    const unsigned short* __restrict__ Yh, const unsigned short* __restrict__ Yl,
    const float* __restrict__ T, unsigned* __restrict__ rowmin)
{
    __shared__ uint4 sXh[2][BM * ROWU4], sXl[2][BM * ROWU4];
    __shared__ uint4 sYh[2][BN * ROWU4], sYl[2][BN * ROWU4];
    __shared__ unsigned smin[BM];

    const int tid   = threadIdx.x;
    const int wave  = tid >> 5;
    const int lane  = tid & 31;
    const int lhalf = lane >> 4;
    const int lrow  = lane & 15;
    const int wm    = wave >> 2;
    const int wn    = wave & 3;

    const int rowM = blockIdx.y * BM;
    const int rowN = blockIdx.x * BN;

    if (tid < BM) smin[tid] = 0xFFFFFFFFu;

    const size_t xoff = (size_t)rowM * DDIM;
    const size_t yoff = (size_t)rowN * DDIM;
    const bool issuer = (wave == 0);

    auto issue_chunk = [&](int kc, int buf) {
        const size_t ko = (size_t)kc * BK;
        tdm_load_tile(Xh + xoff + ko, (unsigned)(uintptr_t)(void*)&sXh[buf][0]);
        tdm_load_tile(Xl + xoff + ko, (unsigned)(uintptr_t)(void*)&sXl[buf][0]);
        tdm_load_tile(Yh + yoff + ko, (unsigned)(uintptr_t)(void*)&sYh[buf][0]);
        tdm_load_tile(Yl + yoff + ko, (unsigned)(uintptr_t)(void*)&sYl[buf][0]);
    };

    v8f acc[4][2];
#pragma unroll
    for (int mt = 0; mt < 4; ++mt)
#pragma unroll
        for (int nt = 0; nt < 2; ++nt)
#pragma unroll
            for (int r = 0; r < 8; ++r) acc[mt][nt][r] = 0.f;

    if (issuer) issue_chunk(0, 0);

#pragma unroll 1
    for (int kc = 0; kc < NCHUNK; ++kc) {
        // prefetch next chunk into other buffer; end-of-iteration barrier of the
        // previous loop trip guarantees everyone is done reading it
        if (issuer) {
            if (kc + 1 < NCHUNK) {
                issue_chunk(kc + 1, (kc + 1) & 1);
                __builtin_amdgcn_s_wait_tensorcnt(4);   // chunk kc landed (in-order)
            } else {
                __builtin_amdgcn_s_wait_tensorcnt(0);
            }
        }
        __syncthreads();                 // all waves see chunk kc in LDS
        wmma_chunk(sXh[kc & 1], sXl[kc & 1], sYh[kc & 1], sYl[kc & 1],
                   wm, wn, lrow, lhalf, acc);
        __syncthreads();                 // reads done -> buffer reusable
    }

    min_epilogue(acc, T, rowmin, smin, rowM, rowN, tid, wm, wn, lrow, lhalf);
}
#endif // HAVE_TDM

// ---- kernel 2c: fallback — in-kernel f32->bf16 conversion (round-1 path) --
__global__ __launch_bounds__(256) void gemm_min_cvt_kernel(
    const float* __restrict__ X, const float* __restrict__ Y,
    const float* __restrict__ T, unsigned* __restrict__ rowmin)
{
    __shared__ uint4 sXh[2][BM * ROWU4], sXl[2][BM * ROWU4];
    __shared__ uint4 sYh[2][BN * ROWU4], sYl[2][BN * ROWU4];
    __shared__ unsigned smin[BM];

    const int tid   = threadIdx.x;
    const int wave  = tid >> 5;
    const int lane  = tid & 31;
    const int lhalf = lane >> 4;
    const int lrow  = lane & 15;
    const int wm    = wave >> 2;
    const int wn    = wave & 3;

    const int rowM = blockIdx.y * BM;
    const int rowN = blockIdx.x * BN;

    if (tid < BM) smin[tid] = 0xFFFFFFFFu;

    const int grow = tid >> 1, ghalf = tid & 1;
    const float* gx = X + (size_t)(rowM + grow) * DDIM + ghalf * 16;
    const float* gy = Y + (size_t)(rowN + grow) * DDIM + ghalf * 16;

    float4 fx[4], fy[4];
    auto gload = [&](int kc) {
        const float4* px = (const float4*)(gx + kc * BK);
        const float4* py = (const float4*)(gy + kc * BK);
#pragma unroll
        for (int j = 0; j < 4; ++j) { fx[j] = px[j]; fy[j] = py[j]; }
    };
    auto cvtstore = [&](int buf) {
        float vx[16], vy[16];
#pragma unroll
        for (int j = 0; j < 4; ++j) {
            vx[4*j+0]=fx[j].x; vx[4*j+1]=fx[j].y; vx[4*j+2]=fx[j].z; vx[4*j+3]=fx[j].w;
            vy[4*j+0]=fy[j].x; vy[4*j+1]=fy[j].y; vy[4*j+2]=fy[j].z; vy[4*j+3]=fy[j].w;
        }
        unsigned hxq[8], lxq[8], hyq[8], lyq[8];
#pragma unroll
        for (int e = 0; e < 8; ++e) {
            unsigned short h0 = f2bf(vx[2*e]), h1 = f2bf(vx[2*e+1]);
            unsigned short l0 = f2bf(vx[2*e]   - bf2f(h0));
            unsigned short l1 = f2bf(vx[2*e+1] - bf2f(h1));
            hxq[e] = pack2(h0, h1); lxq[e] = pack2(l0, l1);
            h0 = f2bf(vy[2*e]); h1 = f2bf(vy[2*e+1]);
            l0 = f2bf(vy[2*e]   - bf2f(h0));
            l1 = f2bf(vy[2*e+1] - bf2f(h1));
            hyq[e] = pack2(h0, h1); lyq[e] = pack2(l0, l1);
        }
        const int sb = grow * ROWU4 + ghalf * 2;
        sXh[buf][sb+0] = make_uint4(hxq[0],hxq[1],hxq[2],hxq[3]);
        sXh[buf][sb+1] = make_uint4(hxq[4],hxq[5],hxq[6],hxq[7]);
        sXl[buf][sb+0] = make_uint4(lxq[0],lxq[1],lxq[2],lxq[3]);
        sXl[buf][sb+1] = make_uint4(lxq[4],lxq[5],lxq[6],lxq[7]);
        sYh[buf][sb+0] = make_uint4(hyq[0],hyq[1],hyq[2],hyq[3]);
        sYh[buf][sb+1] = make_uint4(hyq[4],hyq[5],hyq[6],hyq[7]);
        sYl[buf][sb+0] = make_uint4(lyq[0],lyq[1],lyq[2],lyq[3]);
        sYl[buf][sb+1] = make_uint4(lyq[4],lyq[5],lyq[6],lyq[7]);
    };

    v8f acc[4][2];
#pragma unroll
    for (int mt = 0; mt < 4; ++mt)
#pragma unroll
        for (int nt = 0; nt < 2; ++nt)
#pragma unroll
            for (int r = 0; r < 8; ++r) acc[mt][nt][r] = 0.f;

    gload(0);
    cvtstore(0);
    __syncthreads();
#pragma unroll 1
    for (int kc = 0; kc < NCHUNK; ++kc) {
        if (kc + 1 < NCHUNK) gload(kc + 1);
        if (kc + 2 < NCHUNK) {
            __builtin_prefetch(gx + (kc + 2) * BK, 0, 1);
            __builtin_prefetch(gy + (kc + 2) * BK, 0, 1);
        }
        wmma_chunk(sXh[kc & 1], sXl[kc & 1], sYh[kc & 1], sYl[kc & 1],
                   wm, wn, lrow, lhalf, acc);
        if (kc + 1 < NCHUNK) {
            cvtstore((kc + 1) & 1);
            __syncthreads();
        }
    }

    min_epilogue(acc, T, rowmin, smin, rowM, rowN, tid, wm, wn, lrow, lhalf);
}

// ---- kernel 3: final reduction -------------------------------------------
__global__ __launch_bounds__(256) void finish_kernel(
    const float* __restrict__ x2, const unsigned* __restrict__ rowmin,
    const float* __restrict__ PSI, float* __restrict__ out, int N, int M)
{
    __shared__ float red[256];
    const int tid = threadIdx.x;

    float s = 0.f;
    for (int i = tid; i < N; i += 256) s += x2[i] + ord2f(rowmin[i]);
    red[tid] = s;
    __syncthreads();
#pragma unroll
    for (int w = 128; w >= 1; w >>= 1) {
        if (tid < w) red[tid] += red[tid + w];
        __syncthreads();
    }
    float sum1 = red[0];
    __syncthreads();

    float p = 0.f;
    for (int j = tid; j < M; j += 256) p += PSI[j];
    red[tid] = p;
    __syncthreads();
#pragma unroll
    for (int w = 128; w >= 1; w >>= 1) {
        if (tid < w) red[tid] += red[tid + w];
        __syncthreads();
    }
    if (tid == 0) out[0] = sum1 / (float)N + red[0] / (float)M;
}

// ---------------------------------------------------------------------------

extern "C" void kernel_launch(void* const* d_in, const int* in_sizes, int n_in,
                              void* d_out, int out_size, void* d_ws, size_t ws_size,
                              hipStream_t stream) {
    const float* X   = (const float*)d_in[0];   // [N, 512] f32
    const float* Y   = (const float*)d_in[1];   // [M, 512] f32
    const float* PSI = (const float*)d_in[2];   // [M]      f32
    float* out = (float*)d_out;

    const int N = in_sizes[0] / DDIM;
    const int M = in_sizes[2];

    float*    x2     = (float*)d_ws;
    float*    t      = x2 + N;
    unsigned* rowmin = (unsigned*)(t + M);

    const size_t hdr    = (size_t)(2 * N + M) * 4;            // x2 + t + rowmin
    const size_t planes = (size_t)(N + M) * DDIM * 4;         // hi+lo bf16 for X and Y

    prep_kernel<<<(N + M) / 8, 256, 0, stream>>>(X, Y, PSI, x2, t, rowmin, N, M);

    dim3 grid(M / BN, N / BM);
#if HAVE_TDM
    if (ws_size >= hdr + planes) {
        unsigned short* Xh = (unsigned short*)((char*)d_ws + hdr);
        unsigned short* Xl = Xh + (size_t)N * DDIM;
        unsigned short* Yh = Xl + (size_t)N * DDIM;
        unsigned short* Yl = Yh + (size_t)M * DDIM;
        convert_kernel<<<(int)(((size_t)(N + M) * DDIM) / 2048), 256, 0, stream>>>(
            X, Y, Xh, Xl, Yh, Yl, N);
        gemm_min_tdm_kernel<<<grid, 256, 0, stream>>>(Xh, Xl, Yh, Yl, t, rowmin);
    } else
#endif
    {
        (void)planes; (void)hdr;
        gemm_min_cvt_kernel<<<grid, 256, 0, stream>>>(X, Y, t, rowmin);
    }

    finish_kernel<<<1, 256, 0, stream>>>(x2, rowmin, PSI, out, N, M);
}
